// ScaledDotProductAttention_1262720385467
// MI455X (gfx1250) — matrix operand order repaired and checked by reading, not compiled, not run on hardware
//
#include <hip/hip_runtime.h>
#include <hip/hip_bf16.h>

// ---------------------------------------------------------------------------
// MHA forward for MI455X (gfx1250, wave32, WMMA 16x16x32 f16)
// B=4, NQ=NK=2048, D_MODEL=1024, H=16, DK=DV=64
// Stage 0: fp32 -> f16 conversion of activations + weights (one streaming pass)
// Stage 1: Q/K/V projections, f16 WMMA, 32x64 wave tile, ping-pong k-loop
// Stage 2: flash attention (online softmax, LDS C->A transpose of P)
// Stage 3: output projection, fp32 epilogue
// ---------------------------------------------------------------------------

typedef __attribute__((ext_vector_type(16))) _Float16 v16h;
typedef __attribute__((ext_vector_type(8)))  _Float16 v8h;
typedef __attribute__((ext_vector_type(8)))  float    v8f;

#define NSEQ   2048
#define DMODEL 1024
#define NHEAD  16

__device__ __forceinline__ v8f wmma_f16(v16h a, v16h b, v8f c) {
  return __builtin_amdgcn_wmma_f32_16x16x32_f16(
      /*neg_a=*/false, a, /*neg_b=*/false, b,
      /*c_mod=*/(short)0, c, /*reuse_a=*/false, /*reuse_b=*/false);
}

__device__ __forceinline__ v16h cat16(v8h a, v8h b) {
  return __builtin_shufflevector(a, b, 0, 1, 2, 3, 4, 5, 6, 7,
                                        8, 9, 10, 11, 12, 13, 14, 15);
}

__device__ __forceinline__ v8h ld8h(const _Float16* p) {
  return *reinterpret_cast<const v8h*>(p);
}

// ---------------------------------------------------------------------------
// Stage 0: streaming fp32 -> f16 (8 elements / lane, b128 in, b128 out)
// ---------------------------------------------------------------------------
__global__ __launch_bounds__(256) void cvt_f16_kernel(
    const float* __restrict__ in, _Float16* __restrict__ out, int n8) {
  const int i = blockIdx.x * 256 + threadIdx.x;
  if (i >= n8) return;
  const float4* p = reinterpret_cast<const float4*>(in + (size_t)i * 8);
  const float4 x = p[0], y = p[1];
  v8h o;
  o[0] = (_Float16)x.x; o[1] = (_Float16)x.y;
  o[2] = (_Float16)x.z; o[3] = (_Float16)x.w;
  o[4] = (_Float16)y.x; o[5] = (_Float16)y.y;
  o[6] = (_Float16)y.z; o[7] = (_Float16)y.w;
  *reinterpret_cast<v8h*>(out + (size_t)i * 8) = o;
}

// ---------------------------------------------------------------------------
// Stages 1 & 3: GEMM  C[8192,1024] = A[8192,1024] @ W^T[1024,1024] + bias
//   A, W already f16. One wave owns a 32x64 tile (2 row-blocks x 4 col-tiles)
//   so each B-fragment is reused by two WMMAs. k-loop: 2x-unrolled ping-pong.
//   out_mode: 0 -> f16 [b,h,n,64] ; 1 -> f16 [b,h,64,n] ; 2 -> fp32 row-major
// ---------------------------------------------------------------------------
struct KStep {                 // raw fragments for one 32-wide k-step
  v8h a0, a1;                  // A row-block 0: runs hi*16 , hi*16+8
  v8h a2, a3;                  // A row-block 1
  v8h b0[4], b1[4];            // B run: hi*16 , hi*16+8 (per 16-col tile)
};

__device__ __forceinline__ void load_step(
    KStep& s, const _Float16* ar0, const _Float16* ar1,
    const _Float16* wr0, const _Float16* wr1,
    const _Float16* wr2, const _Float16* wr3, int k, int hi) {
  s.a0 = ld8h(ar0 + k + hi * 16);
  s.a1 = ld8h(ar0 + k + hi * 16 + 8);
  s.a2 = ld8h(ar1 + k + hi * 16);
  s.a3 = ld8h(ar1 + k + hi * 16 + 8);
  s.b0[0] = ld8h(wr0 + k + hi * 16); s.b1[0] = ld8h(wr0 + k + hi * 16 + 8);
  s.b0[1] = ld8h(wr1 + k + hi * 16); s.b1[1] = ld8h(wr1 + k + hi * 16 + 8);
  s.b0[2] = ld8h(wr2 + k + hi * 16); s.b1[2] = ld8h(wr2 + k + hi * 16 + 8);
  s.b0[3] = ld8h(wr3 + k + hi * 16); s.b1[3] = ld8h(wr3 + k + hi * 16 + 8);
}

__device__ __forceinline__ void mma_step(v8f accLo[4], v8f accHi[4],
                                         const KStep& s) {
  const v16h aLo = cat16(s.a0, s.a1);
  const v16h aHi = cat16(s.a2, s.a3);
#pragma unroll
  for (int nb = 0; nb < 4; ++nb) {
    const v16h b = cat16(s.b0[nb], s.b1[nb]);
    accLo[nb] = wmma_f16(aLo, b, accLo[nb]);
    accHi[nb] = wmma_f16(aHi, b, accHi[nb]);
  }
}

__global__ __launch_bounds__(128) void gemm_f16_kernel(
    const _Float16* __restrict__ A, const _Float16* __restrict__ W,
    const float* __restrict__ bias, _Float16* __restrict__ outh,
    float* __restrict__ outf, int out_mode) {
  const int lane = threadIdx.x & 31;
  const int warp = threadIdx.x >> 5;
  const int wid  = blockIdx.x * 4 + warp;
  const int M_TILES = (4 * NSEQ) / 32;      // 256 (32-row wave tiles)
  const int mTile = wid % M_TILES;
  const int nGrp  = wid / M_TILES;          // 0..15
  const int m0 = mTile * 32;
  const int c0 = nGrp * 64;
  const int lr = lane & 15, hi = lane >> 4;

  const _Float16* ar0 = A + (size_t)(m0 + lr) * DMODEL;       // rows m0..m0+15
  const _Float16* ar1 = ar0 + (size_t)16 * DMODEL;            // rows m0+16..+31
  const _Float16* wr0 = W + (size_t)(c0 +  0 + lr) * DMODEL;  // B: col = lr
  const _Float16* wr1 = W + (size_t)(c0 + 16 + lr) * DMODEL;
  const _Float16* wr2 = W + (size_t)(c0 + 32 + lr) * DMODEL;
  const _Float16* wr3 = W + (size_t)(c0 + 48 + lr) * DMODEL;

  v8f accLo[4] = {}, accHi[4] = {};
  KStep s0, s1;
  load_step(s0, ar0, ar1, wr0, wr1, wr2, wr3, 0, hi);

#pragma unroll 1
  for (int k = 0; k < DMODEL - 64; k += 64) {
    load_step(s1, ar0, ar1, wr0, wr1, wr2, wr3, k + 32, hi); // issue before mma
    mma_step(accLo, accHi, s0);
    load_step(s0, ar0, ar1, wr0, wr1, wr2, wr3, k + 64, hi);
    mma_step(accLo, accHi, s1);
  }
  // tail: k = DMODEL-64 (in s0), then k = DMODEL-32
  load_step(s1, ar0, ar1, wr0, wr1, wr2, wr3, DMODEL - 32, hi);
  mma_step(accLo, accHi, s0);
  mma_step(accLo, accHi, s1);

#pragma unroll
  for (int blk = 0; blk < 2; ++blk) {
    const v8f* acc = blk ? accHi : accLo;
#pragma unroll
    for (int nb = 0; nb < 4; ++nb) {
      const int c = c0 + nb * 16 + lr;
      const float bvv = bias[c];
      const int h = c >> 6, d = c & 63;
#pragma unroll
      for (int r = 0; r < 8; ++r) {
        const int m = m0 + blk * 16 + r + 8 * hi;   // C layout: M = r + 8*hi
        const float v = acc[nb][r] + bvv;
        if (out_mode == 2) {
          outf[(size_t)m * DMODEL + c] = v;
        } else {
          const int bb = m >> 11, n = m & (NSEQ - 1);
          const size_t idx = (out_mode == 1)
              ? ((size_t)(bb * NHEAD + h) * 64 + d) * NSEQ + n
              : ((size_t)(bb * NHEAD + h) * NSEQ + n) * 64 + d;
          outh[idx] = (_Float16)v;
        }
      }
    }
  }
}

// ---------------------------------------------------------------------------
// Stage 2: flash attention. One wave per (b, h, 16-query tile).
//   Qh, Kh: [b, h, 2048, 64] f16 ; Vt: [b, h, 64, 2048] f16
//   Oa: [b, n, h*64] f16
// ---------------------------------------------------------------------------
__global__ __launch_bounds__(128) void flash_attn_kernel(
    const _Float16* __restrict__ Qh, const _Float16* __restrict__ Kh,
    const _Float16* __restrict__ Vt, _Float16* __restrict__ Oa) {
  __shared__ __align__(16) _Float16 lds[4 * 16 * 32];   // per-wave 16x32 P slab
  const int lane = threadIdx.x & 31;
  const int warp = threadIdx.x >> 5;
  const int wid  = blockIdx.x * 4 + warp;
  const int bh = wid >> 7;                  // 0..63
  const int q0 = (wid & 127) * 16;
  const int lr = lane & 15, hi = lane >> 4;
  _Float16* slab = lds + warp * (16 * 32);

  const _Float16* qrow = Qh + ((size_t)bh * NSEQ + q0 + lr) * 64;
  const v16h qa0 = cat16(ld8h(qrow + hi * 16),      ld8h(qrow + hi * 16 + 8));
  const v16h qa1 = cat16(ld8h(qrow + 32 + hi * 16), ld8h(qrow + 32 + hi * 16 + 8));

  float mrun[8], srun[8];
#pragma unroll
  for (int r = 0; r < 8; ++r) { mrun[r] = -1e30f; srun[r] = 0.f; }
  v8f acc[4] = {};                          // O tile: 16 q x 64 d

  const _Float16* kb_all = Kh + (size_t)bh * NSEQ * 64;
  const _Float16* vb_all = Vt + (size_t)bh * 64 * NSEQ;

#pragma unroll 1
  for (int kk = 0; kk < NSEQ; kk += 32) {
    // ---- issue ALL loads of this iteration up front (K: 8xb128, V: 8xb128)
    const _Float16* k0r = kb_all + (size_t)(kk + lr) * 64;        // B col = key
    const _Float16* k1r = kb_all + (size_t)(kk + 16 + lr) * 64;
    v8h kraw[4][2];
    kraw[0][0] = ld8h(k0r + hi * 16);      kraw[0][1] = ld8h(k0r + hi * 16 + 8);
    kraw[1][0] = ld8h(k0r + 32 + hi * 16); kraw[1][1] = ld8h(k0r + 32 + hi * 16 + 8);
    kraw[2][0] = ld8h(k1r + hi * 16);      kraw[2][1] = ld8h(k1r + hi * 16 + 8);
    kraw[3][0] = ld8h(k1r + 32 + hi * 16); kraw[3][1] = ld8h(k1r + 32 + hi * 16 + 8);
    v8h vraw[4][2];
#pragma unroll
    for (int t = 0; t < 4; ++t) {
      const _Float16* vr = vb_all + (size_t)(t * 16 + lr) * NSEQ + kk;
      vraw[t][0] = ld8h(vr + hi * 16);
      vraw[t][1] = ld8h(vr + hi * 16 + 8);
    }

    // ---- scores: S0 = keys kk..kk+15, S1 = keys kk+16..kk+31
    v8f s0 = {};
    s0 = wmma_f16(qa0, cat16(kraw[0][0], kraw[0][1]), s0);
    s0 = wmma_f16(qa1, cat16(kraw[1][0], kraw[1][1]), s0);
    v8f s1 = {};
    s1 = wmma_f16(qa0, cat16(kraw[2][0], kraw[2][1]), s1);
    s1 = wmma_f16(qa1, cat16(kraw[3][0], kraw[3][1]), s1);

    // ---- online softmax per query row (row m = r + 8*hi in one 16-lane group)
#pragma unroll
    for (int r = 0; r < 8; ++r) {
      float v0 = s0[r] * 0.125f;            // 1/sqrt(64)
      float v1 = s1[r] * 0.125f;
      float mx = fmaxf(v0, v1);
      mx = fmaxf(mx, __shfl_xor(mx, 1));
      mx = fmaxf(mx, __shfl_xor(mx, 2));
      mx = fmaxf(mx, __shfl_xor(mx, 4));
      mx = fmaxf(mx, __shfl_xor(mx, 8));
      const float mnew = fmaxf(mrun[r], mx);
      const float corr = __expf(mrun[r] - mnew);
      mrun[r] = mnew;
      const float p0 = __expf(v0 - mnew);
      const float p1 = __expf(v1 - mnew);
      float rs = p0 + p1;
      rs += __shfl_xor(rs, 1);
      rs += __shfl_xor(rs, 2);
      rs += __shfl_xor(rs, 4);
      rs += __shfl_xor(rs, 8);
      srun[r] = srun[r] * corr + rs;
#pragma unroll
      for (int t = 0; t < 4; ++t) acc[t][r] *= corr;
      const int row = r + 8 * hi;
      slab[row * 32 + lr]      = (_Float16)p0;   // C-layout -> LDS transpose
      slab[row * 32 + 16 + lr] = (_Float16)p1;
    }
    asm volatile("s_wait_dscnt 0" ::: "memory");  // intra-wave LDS RAW

    // ---- reload P as an A fragment (16 q x 32 keys) and accumulate O
    const _Float16* prow = slab + lr * 32;
  const v16h pa = cat16(ld8h(prow + hi * 16), ld8h(prow + hi * 16 + 8));
#pragma unroll
    for (int t = 0; t < 4; ++t)
      acc[t] = wmma_f16(pa, cat16(vraw[t][0], vraw[t][1]), acc[t]);
  }

  // ---- normalize, emit [b, n, h*64] f16
  const int b = bh >> 4, h = bh & 15;
#pragma unroll
  for (int r = 0; r < 8; ++r) {
    const float inv = 1.0f / srun[r];
    const int q = q0 + r + 8 * hi;
    const size_t base = ((size_t)(b * NSEQ + q)) * DMODEL + h * 64;
#pragma unroll
    for (int t = 0; t < 4; ++t)
      Oa[base + t * 16 + lr] = (_Float16)(acc[t][r] * inv);
  }
}

// ---------------------------------------------------------------------------
extern "C" void kernel_launch(void* const* d_in, const int* in_sizes, int n_in,
                              void* d_out, int out_size, void* d_ws, size_t ws_size,
                              hipStream_t stream) {
  (void)in_sizes; (void)n_in; (void)out_size; (void)ws_size;
  const float* queries = (const float*)d_in[0];
  const float* keys    = (const float*)d_in[1];
  const float* values  = (const float*)d_in[2];
  const float* Wq = (const float*)d_in[3];
  const float* bq = (const float*)d_in[4];
  const float* Wk = (const float*)d_in[5];
  const float* bk = (const float*)d_in[6];
  const float* Wv = (const float*)d_in[7];
  const float* bv = (const float*)d_in[8];
  const float* Wo = (const float*)d_in[9];
  const float* bo = (const float*)d_in[10];
  float* out = (float*)d_out;

  const size_t SEG  = (size_t)4 * NHEAD * NSEQ * 64;  // 8,388,608 halves (16 MB)
  const size_t WSEG = (size_t)DMODEL * DMODEL;        // 1,048,576 halves (2 MB)
  _Float16* XhQ = (_Float16*)d_ws;                    // converted activations
  _Float16* XhK = XhQ + SEG;
  _Float16* XhV = XhK + SEG;
  _Float16* WhQ = XhV + SEG;                          // converted weights
  _Float16* WhK = WhQ + WSEG;
  _Float16* WhV = WhK + WSEG;
  _Float16* WhO = WhV + WSEG;
  _Float16* Qh  = WhO + WSEG;                         // projected heads
  _Float16* Kh  = Qh + SEG;
  _Float16* Vt  = Kh + SEG;
  _Float16* Ao  = XhQ;   // alias: XhQ is dead once the Q projection has run

  dim3 b256(256), b128(128);
  const int ACT8 = (int)(SEG / 8);    // 1,048,576 -> 4096 blocks
  const int W8   = (int)(WSEG / 8);   //   131,072 ->  512 blocks

  // Stage 0: one streaming conversion pass
  cvt_f16_kernel<<<ACT8 / 256, b256, 0, stream>>>(queries, XhQ, ACT8);
  cvt_f16_kernel<<<ACT8 / 256, b256, 0, stream>>>(keys,    XhK, ACT8);
  cvt_f16_kernel<<<ACT8 / 256, b256, 0, stream>>>(values,  XhV, ACT8);
  cvt_f16_kernel<<<W8 / 256,   b256, 0, stream>>>(Wq, WhQ, W8);
  cvt_f16_kernel<<<W8 / 256,   b256, 0, stream>>>(Wk, WhK, W8);
  cvt_f16_kernel<<<W8 / 256,   b256, 0, stream>>>(Wv, WhV, W8);
  cvt_f16_kernel<<<W8 / 256,   b256, 0, stream>>>(Wo, WhO, W8);

  // Stage 1: projections (256 m-tiles x 16 n-groups = 4096 waves = 1024 blocks)
  gemm_f16_kernel<<<1024, b128, 0, stream>>>(XhQ, WhQ, bq, Qh, nullptr, 0);
  gemm_f16_kernel<<<1024, b128, 0, stream>>>(XhK, WhK, bk, Kh, nullptr, 0);
  gemm_f16_kernel<<<1024, b128, 0, stream>>>(XhV, WhV, bv, Vt, nullptr, 1);

  // Stage 2: attention (64 bh x 128 q-tiles = 8192 waves = 2048 blocks)
  flash_attn_kernel<<<2048, b128, 0, stream>>>(Qh, Kh, Vt, Ao);

  // Stage 3: output projection, fp32 epilogue into d_out
  gemm_f16_kernel<<<1024, b128, 0, stream>>>(Ao, WhO, bo, nullptr, out, 2);
}
